// GaussianProcessEmbeddingHead_59605556134142
// MI455X (gfx1250) — matrix operand order repaired and checked by reading, not compiled, not run on hardware
//
#include <hip/hip_runtime.h>
#include <hip/hip_bf16.h>

// GaussianProcessEmbeddingHead, CDNA5/gfx1250.
//
// Algebraic optimization: diag(K_rbf) == exp(0) == 1 exactly, so
// sigma_adjusted == sigma; the O(B*N^2*D) Gram matrix is dead work -> skipped.
//
// Precision/throughput choice: bf16x3 split GEMM (x = hi + lo; keep
// hi*hi + hi*lo + lo*hi, fp32 accumulate) on V_WMMA_F32_16X16X32_BF16.
// 32 K-values cost 3 bf16 WMMAs vs 8 fp32 16x16x4 WMMAs (~2.7x fewer
// matrix-pipe cycles at the ~8x-faster bf16 rate), with fp32-comparable
// accuracy (dropped lo*lo term ~2^-18 relative per product). Split
// conversions are VALU ops that co-execute with the 16-bit WMMA pipe.

typedef __attribute__((ext_vector_type(16))) __bf16 v16bf;
typedef __attribute__((ext_vector_type(8)))  __bf16 v8bf;
typedef __attribute__((ext_vector_type(4)))  __bf16 v4bf;
typedef __attribute__((ext_vector_type(8)))  float  v8f;

#define M_TOT 16384   // B*N = 8*2048
#define K_TOT 1024    // D_in
#define E_TOT 512     // D_emb
#define BM    128     // block tile M
#define BE    64      // block tile E
#define BK    64      // K tile staged through LDS (two 32-wide WMMA chunks)
#define LDB   72      // padded LDS row stride in bf16 (144B: 16B-aligned frags,
                      // lane bank starts spaced exactly 4 dwords -> no conflicts)

// Load a v16bf fragment as two contiguous 8x-bf16 (16B) LDS reads.
__device__ __forceinline__ v16bf load_frag2(const __bf16* p, int elem_gap) {
    union { v16bf v; v8bf h[2]; } u;
    u.h[0] = *reinterpret_cast<const v8bf*>(p);
    u.h[1] = *reinterpret_cast<const v8bf*>(p + elem_gap);
    return u.v;
}

// Split a float4 into bf16 hi + bf16 residual lo.
__device__ __forceinline__ void split4(const float4 v, v4bf& hi, v4bf& lo) {
    const float f[4] = {v.x, v.y, v.z, v.w};
#pragma unroll
    for (int i = 0; i < 4; ++i) {
        const __bf16 h = (__bf16)f[i];
        hi[i] = h;
        lo[i] = (__bf16)(f[i] - (float)h);
    }
}

__global__ __launch_bounds__(256, 2)
void gpe_dual_head_wmma_bf16x3(const float* __restrict__ x,
                               const float* __restrict__ W_mu,
                               const float* __restrict__ b_mu,
                               const float* __restrict__ W_lv,
                               const float* __restrict__ b_lv,
                               float* __restrict__ out_mu,
                               float* __restrict__ out_sig) {
    __shared__ __align__(16) __bf16 sXhi[BM * LDB];   // 18432 B
    __shared__ __align__(16) __bf16 sXlo[BM * LDB];   // 18432 B
    __shared__ __align__(16) __bf16 sWmh[BE * LDB];   //  9216 B
    __shared__ __align__(16) __bf16 sWml[BE * LDB];
    __shared__ __align__(16) __bf16 sWvh[BE * LDB];
    __shared__ __align__(16) __bf16 sWvl[BE * LDB];   // total 73728 B

    const int tid  = threadIdx.x;         // 0..255
    const int lane = tid & 31;            // wave32 lane
    const int wave = tid >> 5;            // 0..7
    const int wm   = wave >> 1;           // 0..3  (M direction)
    const int we   = wave & 1;            // 0..1  (E direction)

    const int m0 = blockIdx.x * BM;       // 128 blocks in M
    const int e0 = blockIdx.y * BE;       // 8 blocks in E

    // Staging: 64 cols = 16 float4 per row -> 16 threads per row.
    const int ldRow = tid >> 4;           // 0..15
    const int ldCol = (tid & 15) << 2;    // 0,4,...,60

    // 16-bit WMMA fragment addressing (wave32):
    //  A 16x32: lanes 0-15 hold K=0..15, lanes 16-31 K=16..31 (same order as B).
    //  B 32x16 (sparse-B analogous): lanes 0-15 hold K=0..15, lanes 16-31 K=16..31.
    const int fr  = lane & 15;
    const int ka8 = (lane >> 4) << 3;     // unused 8-element half offset; A frag uses kb too
    const int kb  = (lane >> 4) << 4;     // B frag: 16-element half offset

    v8f accMu[2][2];  // [m_sub][e_sub]
    v8f accLv[2][2];
#pragma unroll
    for (int ms = 0; ms < 2; ++ms)
#pragma unroll
        for (int es = 0; es < 2; ++es) {
            accMu[ms][es] = {};
            accLv[ms][es] = {};
        }

    for (int kt = 0; kt < K_TOT; kt += BK) {
        __syncthreads();  // protect LDS reuse across iterations

        // ---- stage x tile: 128 x 64 fp32 -> bf16 hi/lo (8 float4/thread) ----
#pragma unroll
        for (int i = 0; i < 8; ++i) {
            const int r = ldRow + i * 16;
            const float4 v = *reinterpret_cast<const float4*>(
                x + (size_t)(m0 + r) * K_TOT + kt + ldCol);
            v4bf hi, lo;
            split4(v, hi, lo);
            *reinterpret_cast<v4bf*>(&sXhi[r * LDB + ldCol]) = hi;
            *reinterpret_cast<v4bf*>(&sXlo[r * LDB + ldCol]) = lo;
        }
        // ---- stage W_mu / W_logvar tiles: 64 x 64 each ----
#pragma unroll
        for (int i = 0; i < 4; ++i) {
            const int r = ldRow + i * 16;
            const size_t goff = (size_t)(e0 + r) * K_TOT + kt + ldCol;
            v4bf hi, lo;
            const float4 vm = *reinterpret_cast<const float4*>(W_mu + goff);
            split4(vm, hi, lo);
            *reinterpret_cast<v4bf*>(&sWmh[r * LDB + ldCol]) = hi;
            *reinterpret_cast<v4bf*>(&sWml[r * LDB + ldCol]) = lo;
            const float4 vv = *reinterpret_cast<const float4*>(W_lv + goff);
            split4(vv, hi, lo);
            *reinterpret_cast<v4bf*>(&sWvh[r * LDB + ldCol]) = hi;
            *reinterpret_cast<v4bf*>(&sWvl[r * LDB + ldCol]) = lo;
        }
        // ---- prefetch next K tile into cache (global_prefetch_b8) ----
        if (kt + BK < K_TOT) {
            __builtin_prefetch(x + (size_t)(m0 + ldRow) * K_TOT + kt + BK + ldCol, 0, 0);
            __builtin_prefetch(W_mu + (size_t)(e0 + ldRow) * K_TOT + kt + BK + ldCol, 0, 0);
            __builtin_prefetch(W_lv + (size_t)(e0 + ldRow) * K_TOT + kt + BK + ldCol, 0, 0);
        }
        __syncthreads();

        // ---- compute: two 32-wide chunks, 24 bf16 WMMAs each ----
#pragma unroll
        for (int kc = 0; kc < BK; kc += 32) {
            v16bf ah[2], al[2];
#pragma unroll
            for (int ms = 0; ms < 2; ++ms) {
                const int base = (wm * 32 + ms * 16 + fr) * LDB + kc + kb;
                ah[ms] = load_frag2(&sXhi[base], 8);
                al[ms] = load_frag2(&sXlo[base], 8);
            }
            v16bf bmh[2], bml[2], bvh[2], bvl[2];
#pragma unroll
            for (int es = 0; es < 2; ++es) {
                const int base = (we * 32 + es * 16 + fr) * LDB + kc + kb;
                bmh[es] = load_frag2(&sWmh[base], 8);
                bml[es] = load_frag2(&sWml[base], 8);
                bvh[es] = load_frag2(&sWvh[base], 8);
                bvl[es] = load_frag2(&sWvl[base], 8);
            }
#pragma unroll
            for (int ms = 0; ms < 2; ++ms)
#pragma unroll
                for (int es = 0; es < 2; ++es) {
                    // mu head: hi*hi + hi*lo + lo*hi
                    accMu[ms][es] = __builtin_amdgcn_wmma_f32_16x16x32_bf16(
                        false, ah[ms], false, bmh[es], (short)0, accMu[ms][es], false, false);
                    accLv[ms][es] = __builtin_amdgcn_wmma_f32_16x16x32_bf16(
                        false, ah[ms], false, bvh[es], (short)0, accLv[ms][es], false, false);
                    accMu[ms][es] = __builtin_amdgcn_wmma_f32_16x16x32_bf16(
                        false, ah[ms], false, bml[es], (short)0, accMu[ms][es], false, false);
                    accLv[ms][es] = __builtin_amdgcn_wmma_f32_16x16x32_bf16(
                        false, ah[ms], false, bvl[es], (short)0, accLv[ms][es], false, false);
                    accMu[ms][es] = __builtin_amdgcn_wmma_f32_16x16x32_bf16(
                        false, al[ms], false, bmh[es], (short)0, accMu[ms][es], false, false);
                    accLv[ms][es] = __builtin_amdgcn_wmma_f32_16x16x32_bf16(
                        false, al[ms], false, bvh[es], (short)0, accLv[ms][es], false, false);
                }
        }
    }

    // ---- epilogue: bias add, sigma = exp(0.5*logvar) * cov_diag(==1) ----
    const float HALF_LOG2E = 0.72134752044448170368f;  // 0.5 * log2(e)
    float bmu[2], blv[2];
    int   eIdx[2];
#pragma unroll
    for (int es = 0; es < 2; ++es) {
        const int e = e0 + we * 32 + es * 16 + (lane & 15);
        eIdx[es] = e;
        bmu[es]  = b_mu[e];
        blv[es]  = b_lv[e];
    }
#pragma unroll
    for (int ms = 0; ms < 2; ++ms) {
        const int mBase = m0 + wm * 32 + ms * 16 + ((lane >> 4) << 3);
#pragma unroll
        for (int es = 0; es < 2; ++es) {
#pragma unroll
            for (int vi = 0; vi < 8; ++vi) {
                const size_t idx = (size_t)(mBase + vi) * E_TOT + eIdx[es];
                out_mu[idx]  = accMu[ms][es][vi] + bmu[es];
                out_sig[idx] = exp2f((accLv[ms][es][vi] + blv[es]) * HALF_LOG2E);
            }
        }
    }
}

extern "C" void kernel_launch(void* const* d_in, const int* in_sizes, int n_in,
                              void* d_out, int out_size, void* d_ws, size_t ws_size,
                              hipStream_t stream) {
    (void)in_sizes; (void)n_in; (void)d_ws; (void)ws_size; (void)out_size;
    const float* x    = (const float*)d_in[0];
    const float* W_mu = (const float*)d_in[1];
    const float* b_mu = (const float*)d_in[2];
    const float* W_lv = (const float*)d_in[3];
    const float* b_lv = (const float*)d_in[4];
    float* out_mu  = (float*)d_out;
    float* out_sig = (float*)d_out + (size_t)M_TOT * E_TOT;

    dim3 grid(M_TOT / BM, E_TOT / BE);  // (128, 8)
    dim3 block(256);                    // 8 wave32 waves
    gpe_dual_head_wmma_bf16x3<<<grid, block, 0, stream>>>(x, W_mu, b_mu, W_lv, b_lv,
                                                          out_mu, out_sig);
}
